// MultiheadSelfAttention_48326972014756
// MI455X (gfx1250) — compile-verified, matrix operand order checked by reading, not run on hardware
//
#include <hip/hip_runtime.h>
#include <hip/hip_bf16.h>

typedef __bf16 bf16_t;
typedef __attribute__((ext_vector_type(16))) __bf16 v16bf;
typedef __attribute__((ext_vector_type(8)))  float  v8f;
typedef __attribute__((ext_vector_type(4)))  unsigned int v4u;
typedef __attribute__((ext_vector_type(4)))  int  v4i;
typedef __attribute__((ext_vector_type(8)))  int  v8i;

#define WMMA_BF16(a, b, c) \
  __builtin_amdgcn_wmma_f32_16x16x32_bf16(false, (a), false, (b), (short)0, (c), false, false)

#if defined(__gfx1250__) && __has_builtin(__builtin_amdgcn_tensor_load_to_lds) && \
    __has_builtin(__builtin_amdgcn_s_wait_tensorcnt)
#define USE_TDM 1
#else
#define USE_TDM 0
#endif

// f32 -> bf16 round-to-nearest-even via integer ops (no reliance on __bf16 arithmetic)
__device__ __forceinline__ bf16_t f2bf(float f) {
  unsigned u = __builtin_bit_cast(unsigned, f);
  unsigned short r = (unsigned short)((u + 0x7FFFu + ((u >> 16) & 1u)) >> 16);
  return __builtin_bit_cast(bf16_t, r);
}

union Frag   { v16bf v; uint4 q[2]; };
union Chunk8 { uint4 q; bf16_t h[8]; };

// Load one 16-bit WMMA fragment from LDS following the CDNA5 16-bit A/B layout:
// lane holds row (or col) = lane&15, k-base = (lane>>4)*8, with elements
// k = base..base+7 and base+16..base+23  (two 16-byte chunks).
__device__ __forceinline__ v16bf ld_frag(const bf16_t* p) {
  Frag f;
  f.q[0] = *(const uint4*)(p);
  f.q[1] = *(const uint4*)(p + 16);
  return f.v;
}

#if USE_TDM
// Issue a TDM 2D tile load (bf16 elements) global -> LDS.
// Builds D# group0/group1 per CDNA5 ISA ch.8; groups 2/3 zero (2D tensor).
// pad_icode: pad after 2^(pad_icode+1) DWORDs; pad_acode: pad (pad_acode+1) DWORDs.
// This reproduces the skewed LDS pitches used for bank-conflict avoidance.
__device__ __forceinline__ void tdm_load_2d(
    unsigned lds_byte_addr, const void* gaddr,
    unsigned tensor_d0, unsigned tensor_d1,   // tensor extent (elements)
    unsigned tile_d0,   unsigned tile_d1,     // tile extent (elements)
    unsigned stride_d0,                       // tensor row pitch (elements)
    unsigned pad_icode, unsigned pad_acode) {
  unsigned long long ga = (unsigned long long)(size_t)gaddr;
  v4u g0;
  g0[0] = 1u;                                            // count=1, user descriptor
  g0[1] = lds_byte_addr;                                 // lds_addr [63:32]
  g0[2] = (unsigned)(ga & 0xFFFFFFFFu);                  // global_addr lo
  g0[3] = (unsigned)((ga >> 32) & 0x01FFFFFFu) | (2u << 30);  // addr hi | type=2
  v8i g1;
  g1[0] = (int)((1u << 16)                               // data_size = 1 (2 bytes)
              | (1u << 20)                               // pad_enable
              | (pad_icode << 22) | (pad_acode << 25));
  g1[1] = (int)((tensor_d0 & 0xFFFFu) << 16);            // tensor_dim0 lo16 @ [63:48]
  g1[2] = (int)((tensor_d0 >> 16) | ((tensor_d1 & 0xFFFFu) << 16));
  g1[3] = (int)((tensor_d1 >> 16) | (tile_d0 << 16));    // tile_dim0 @ [127:112]
  g1[4] = (int)(tile_d1 & 0xFFFFu);                      // tile_dim1; tile_dim2=0
  g1[5] = (int)stride_d0;                                // tensor_dim0_stride lo32
  g1[6] = 0;                                             // stride hi16 | dim1_stride lo16
  g1[7] = 0;
  v4i gz = {0, 0, 0, 0};
#if __clang_major__ >= 23
  v8i gz8 = {0, 0, 0, 0, 0, 0, 0, 0};
  __builtin_amdgcn_tensor_load_to_lds(g0, g1, gz, gz, gz8, 0);
#else
  __builtin_amdgcn_tensor_load_to_lds(g0, g1, gz, gz, 0);
#endif
}
#endif  // USE_TDM

// ---------------------------------------------------------------------------
// f32 -> bf16 conversion, 4 elements/thread
// ---------------------------------------------------------------------------
__global__ __launch_bounds__(256) void cvt_f32_bf16(const float* __restrict__ src,
                                                    bf16_t* __restrict__ dst, int n4) {
  int i = blockIdx.x * blockDim.x + threadIdx.x;
  if (i < n4) {
    float4 f = ((const float4*)src)[i];
    union { bf16_t h[4]; uint2 v; } o;
    o.h[0] = f2bf(f.x); o.h[1] = f2bf(f.y); o.h[2] = f2bf(f.z); o.h[3] = f2bf(f.w);
    ((uint2*)dst)[i] = o.v;
  }
}

// ---------------------------------------------------------------------------
// C[M,N] = A[M,K] (bf16, row-major) * Bt[N,K]^T  (Bt row-major == B col-major)
// Tile: 128(M) x 64(N) per 256-thread block; 8 waves x (16x64); K slab = 64.
// Double-buffered LDS tiles; TDM (wave 0) prefetches stage s+1 while all waves
// run WMMA on stage s (TENSORcnt is in-order: wait<=2 ==> stage s complete).
// ---------------------------------------------------------------------------
template<bool OUT_BF16>
__global__ __launch_bounds__(256) void gemm_bf16_kernel(
    const bf16_t* __restrict__ A, const bf16_t* __restrict__ Bt,
    void* __restrict__ Cout, int M, int N, int K) {
  constexpr int TP = 72;                     // tile pitch in bf16 (144B, 16B-aligned)
  __shared__ __align__(16) bf16_t As[2][128 * TP];
  __shared__ __align__(16) bf16_t Bs[2][64 * TP];
  const int tid  = threadIdx.x;
  const int wave = tid >> 5, lane = tid & 31;
  const int hi   = lane >> 4, ln = lane & 15;
  const int m0   = blockIdx.y * 128;
  const int n0   = blockIdx.x * 64;

  v8f acc[4] = {};
  const int nStages = K / 64;

#if USE_TDM
  if (wave == 0) {   // wave-uniform branch: TDM issues regardless of EXEC
    // row = 32 DWORDs -> pad_interval code 4; pad 4 DW (8 bf16) -> pitch 72
    tdm_load_2d((unsigned)(size_t)&As[0][0], &A[(size_t)m0 * K],
                (unsigned)K, (unsigned)M, 64u, 128u, (unsigned)K, 4u, 3u);
    tdm_load_2d((unsigned)(size_t)&Bs[0][0], &Bt[(size_t)n0 * K],
                (unsigned)K, (unsigned)N, 64u, 64u, (unsigned)K, 4u, 3u);
  }
#endif

  for (int s = 0; s < nStages; ++s) {
    const int buf = s & 1;
    const int k0 = s * 64;
    __syncthreads();                         // all waves done with buffer buf^1
#if USE_TDM
    if (wave == 0) {
      if (s + 1 < nStages) {
        // prefetch next slab into the other buffer; overlaps with this stage's WMMA
        tdm_load_2d((unsigned)(size_t)&As[buf ^ 1][0], &A[(size_t)m0 * K + k0 + 64],
                    (unsigned)K, (unsigned)M, 64u, 128u, (unsigned)K, 4u, 3u);
        tdm_load_2d((unsigned)(size_t)&Bs[buf ^ 1][0], &Bt[(size_t)n0 * K + k0 + 64],
                    (unsigned)K, (unsigned)N, 64u, 64u, (unsigned)K, 4u, 3u);
        __builtin_amdgcn_s_wait_tensorcnt(2);   // stage s done; s+1 still in flight
      } else {
        __builtin_amdgcn_s_wait_tensorcnt(0);
      }
    }
#else
    // A tile: 128x64 = 1024 chunks of 8 bf16; 4 per thread
    #pragma unroll
    for (int it = 0; it < 4; ++it) {
      int cid = tid + it * 256;
      int r = cid >> 3, kc = (cid & 7) * 8;
      *(uint4*)&As[buf][r * TP + kc] = *(const uint4*)&A[(size_t)(m0 + r) * K + k0 + kc];
    }
    // B tile: 64x64 = 512 chunks; 2 per thread
    #pragma unroll
    for (int it = 0; it < 2; ++it) {
      int cid = tid + it * 256;
      int r = cid >> 3, kc = (cid & 7) * 8;
      *(uint4*)&Bs[buf][r * TP + kc] = *(const uint4*)&Bt[(size_t)(n0 + r) * K + k0 + kc];
    }
    if (k0 + 64 < K)
      __builtin_prefetch(&A[(size_t)(m0 + (tid >> 1)) * K + k0 + 64], 0, 1);
#endif
    __syncthreads();

    #pragma unroll
    for (int kc = 0; kc < 2; ++kc) {
      v16bf a = ld_frag(&As[buf][(wave * 16 + ln) * TP + kc * 32 + hi * 8]);
      #pragma unroll
      for (int n = 0; n < 4; ++n) {
        v16bf b = ld_frag(&Bs[buf][(n * 16 + ln) * TP + kc * 32 + hi * 8]);
        acc[n] = WMMA_BF16(a, b, acc[n]);
      }
    }
  }

  // C layout: lane -> col = ln, rows = hi*8 + j  (VGPR j)
  const int rowBase = m0 + wave * 16 + hi * 8;
  #pragma unroll
  for (int n = 0; n < 4; ++n) {
    int col = n0 + n * 16 + ln;
    #pragma unroll
    for (int j = 0; j < 8; ++j) {
      int row = rowBase + j;
      if constexpr (OUT_BF16)
        ((bf16_t*)Cout)[(size_t)row * N + col] = f2bf(acc[n][j]);
      else
        ((float*)Cout)[(size_t)row * N + col] = acc[n][j];
    }
  }
}

// ---------------------------------------------------------------------------
// Flash attention (causal).  grid = (S/128, H, B), block = 256 (8 waves).
// Each wave owns 16 query rows; k processed in 64-row tiles up to diagonal.
// Q + double-buffered K tiles staged by TDM (pitch 72 via pad); next K tile's
// DMA overlaps current tile's softmax/WMMA. V transposed manually (TDM cannot).
// ---------------------------------------------------------------------------
__global__ __launch_bounds__(256) void attn_kernel(
    const bf16_t* __restrict__ Qb, const bf16_t* __restrict__ Kb,
    const bf16_t* __restrict__ Vb, bf16_t* __restrict__ Ob) {
  constexpr int S = 2048, D = 1024, DK = 64, P = 72;   // pitch 72 bf16 = 144B
  __shared__ __align__(16) bf16_t Qs[128 * P];
  __shared__ __align__(16) bf16_t Ks[2][64 * P];       // double-buffered K tiles
  __shared__ __align__(16) bf16_t Vt[64 * P];          // V transposed: [d][k]
  __shared__ __align__(16) bf16_t Ps[8 * 16 * P];      // per-wave P staging

  const int tid = threadIdx.x, wave = tid >> 5, lane = tid & 31;
  const int hi = lane >> 4, ln = lane & 15;
  const int qi = blockIdx.x, h = blockIdx.y, b = blockIdx.z;
  const int q0 = qi * 128;

  const bf16_t* Qh = Qb + (size_t)b * S * D + (size_t)h * DK;
  const bf16_t* Kh = Kb + (size_t)b * S * D + (size_t)h * DK;
  const bf16_t* Vh = Vb + (size_t)b * S * D + (size_t)h * DK;

  const int ktMax = 2 * qi + 1;                 // causal: only tiles touching <= q0+127

  // Q tile (128x64) loaded once; K tile 0 issued behind it (in-order completion)
#if USE_TDM
  if (wave == 0) {
    // row = 32 DWORDs -> pad_interval code 4; pad 4 DW (8 bf16) -> pitch 72
    tdm_load_2d((unsigned)(size_t)&Qs[0], Qh + (size_t)q0 * D,
                (unsigned)DK, (unsigned)S, 64u, 128u, (unsigned)D, 4u, 3u);
    tdm_load_2d((unsigned)(size_t)&Ks[0][0], Kh,
                (unsigned)DK, (unsigned)S, 64u, 64u, (unsigned)D, 4u, 3u);
    __builtin_amdgcn_s_wait_tensorcnt(1);       // Q complete; K0 may be in flight
  }
#else
  #pragma unroll
  for (int it = 0; it < 4; ++it) {
    int cid = tid + it * 256;
    int r = cid >> 3, kc = (cid & 7) * 8;
    *(uint4*)&Qs[r * P + kc] = *(const uint4*)&Qh[(size_t)(q0 + r) * D + kc];
  }
#endif
  __syncthreads();
  v16bf qf[2];
  #pragma unroll
  for (int kc = 0; kc < 2; ++kc)
    qf[kc] = ld_frag(&Qs[(wave * 16 + ln) * P + kc * 32 + hi * 8]);

  float m[8], l[8];
  v8f ctx[4] = {};
  #pragma unroll
  for (int j = 0; j < 8; ++j) { m[j] = -1e30f; l[j] = 0.f; }

  for (int kt = 0; kt <= ktMax; ++kt) {
    const int k0 = kt * 64;
    const int buf = kt & 1;
    __syncthreads();                            // done with Vt and Ks[buf^1]
#if USE_TDM
    const bool havePrefetch = (kt + 1 <= ktMax);
    if (wave == 0 && havePrefetch)              // prefetch next K tile into other buf
      tdm_load_2d((unsigned)(size_t)&Ks[buf ^ 1][0], Kh + (size_t)(k0 + 64) * D,
                  (unsigned)DK, (unsigned)S, 64u, 64u, (unsigned)D, 4u, 3u);
    // V tile transposed into LDS by all waves, overlapping the K-tile DMA
    #pragma unroll
    for (int it = 0; it < 2; ++it) {
      int cid = tid + it * 256;
      int r = cid >> 3, kc = (cid & 7) * 8;
      Chunk8 c8; c8.q = *(const uint4*)&Vh[(size_t)(k0 + r) * D + kc];
      #pragma unroll
      for (int i = 0; i < 8; ++i) Vt[(kc + i) * P + r] = c8.h[i];
    }
    if (wave == 0) {                            // tile kt complete (in-order)
      if (havePrefetch) __builtin_amdgcn_s_wait_tensorcnt(1);
      else              __builtin_amdgcn_s_wait_tensorcnt(0);
    }
#else
    #pragma unroll
    for (int it = 0; it < 2; ++it) {
      int cid = tid + it * 256;
      int r = cid >> 3, kc = (cid & 7) * 8;
      *(uint4*)&Ks[buf][r * P + kc] = *(const uint4*)&Kh[(size_t)(k0 + r) * D + kc];
      Chunk8 c8; c8.q = *(const uint4*)&Vh[(size_t)(k0 + r) * D + kc];
      #pragma unroll
      for (int i = 0; i < 8; ++i) Vt[(kc + i) * P + r] = c8.h[i];
    }
#endif
    __syncthreads();

    // S = Q * K^T   (B-fragment column n = K row n: contiguous in d)
    v8f sc[4];
    #pragma unroll
    for (int n = 0; n < 4; ++n) {
      v8f c = {};
      #pragma unroll
      for (int kc = 0; kc < 2; ++kc) {
        v16bf bk = ld_frag(&Ks[buf][(n * 16 + ln) * P + kc * 32 + hi * 8]);
        c = WMMA_BF16(qf[kc], bk, c);
      }
      sc[n] = c;
    }

    // scale + causal mask
    const int rowBase = q0 + wave * 16 + hi * 8;
    #pragma unroll
    for (int n = 0; n < 4; ++n) {
      int col = k0 + n * 16 + ln;
      #pragma unroll
      for (int j = 0; j < 8; ++j) {
        float s = sc[n][j] * 0.125f;            // 1/sqrt(64)
        sc[n][j] = (col > rowBase + j) ? -1e30f : s;
      }
    }

    // online softmax: row reductions across the 16 lanes of each half-wave
    #pragma unroll
    for (int j = 0; j < 8; ++j) {
      float v = fmaxf(fmaxf(sc[0][j], sc[1][j]), fmaxf(sc[2][j], sc[3][j]));
      #pragma unroll
      for (int msk = 1; msk < 16; msk <<= 1)
        v = fmaxf(v, __shfl_xor(v, msk, 32));
      float mn = fmaxf(m[j], v);
      float alpha = __expf(m[j] - mn);
      float sum = 0.f;
      #pragma unroll
      for (int n = 0; n < 4; ++n) {
        float pj = __expf(sc[n][j] - mn);
        sc[n][j] = pj;
        sum += pj;
      }
      #pragma unroll
      for (int msk = 1; msk < 16; msk <<= 1)
        sum += __shfl_xor(sum, msk, 32);
      l[j] = l[j] * alpha + sum;
      m[j] = mn;
      #pragma unroll
      for (int n = 0; n < 4; ++n) ctx[n][j] *= alpha;
    }

    // stage P (C-layout -> row-major bf16 in per-wave LDS pad; DS in-order per wave)
    bf16_t* pw = &Ps[wave * 16 * P];
    #pragma unroll
    for (int n = 0; n < 4; ++n)
      #pragma unroll
      for (int j = 0; j < 8; ++j)
        pw[(hi * 8 + j) * P + n * 16 + ln] = f2bf(sc[n][j]);

    // ctx += P * V   (A = P fragments, B-fragment column d = Vt row d)
    #pragma unroll
    for (int kc = 0; kc < 2; ++kc) {
      v16bf a = ld_frag(&pw[ln * P + kc * 32 + hi * 8]);
      #pragma unroll
      for (int n = 0; n < 4; ++n) {
        v16bf bv = ld_frag(&Vt[(n * 16 + ln) * P + kc * 32 + hi * 8]);
        ctx[n] = WMMA_BF16(a, bv, ctx[n]);
      }
    }
  }

  // epilogue: ctx / l  -> bf16 context buffer [B*S, D] at head column h*64
  const int rowBase = b * S + q0 + wave * 16 + hi * 8;
  #pragma unroll
  for (int n = 0; n < 4; ++n) {
    int col = h * DK + n * 16 + ln;
    #pragma unroll
    for (int j = 0; j < 8; ++j) {
      float inv = 1.f / l[j];
      Ob[(size_t)(rowBase + j) * D + col] = f2bf(ctx[n][j] * inv);
    }
  }
}

// ---------------------------------------------------------------------------
extern "C" void kernel_launch(void* const* d_in, const int* in_sizes, int n_in,
                              void* d_out, int out_size, void* d_ws, size_t ws_size,
                              hipStream_t stream) {
  (void)in_sizes; (void)n_in; (void)out_size; (void)ws_size;
  const float* x  = (const float*)d_in[0];
  const float* Wq = (const float*)d_in[1];
  const float* Wk = (const float*)d_in[2];
  const float* Wv = (const float*)d_in[3];
  const float* Wo = (const float*)d_in[4];

  constexpr int B = 4, S = 2048, D = 1024, H = 16;
  constexpr size_t NX = (size_t)B * S * D;   // 8388608
  constexpr size_t NW = (size_t)D * D;       // 1048576

  bf16_t* p   = (bf16_t*)d_ws;
  bf16_t* xb  = p; p += NX;
  bf16_t* wqb = p; p += NW;
  bf16_t* wkb = p; p += NW;
  bf16_t* wvb = p; p += NW;
  bf16_t* wob = p; p += NW;
  bf16_t* qb  = p; p += NX;
  bf16_t* kb  = p; p += NX;
  bf16_t* vb  = p; p += NX;
  bf16_t* cb  = p; p += NX;

  cvt_f32_bf16<<<(int)(NX / 4 / 256), 256, 0, stream>>>(x,  xb,  (int)(NX / 4));
  cvt_f32_bf16<<<(int)(NW / 4 / 256), 256, 0, stream>>>(Wq, wqb, (int)(NW / 4));
  cvt_f32_bf16<<<(int)(NW / 4 / 256), 256, 0, stream>>>(Wk, wkb, (int)(NW / 4));
  cvt_f32_bf16<<<(int)(NW / 4 / 256), 256, 0, stream>>>(Wv, wvb, (int)(NW / 4));
  cvt_f32_bf16<<<(int)(NW / 4 / 256), 256, 0, stream>>>(Wo, wob, (int)(NW / 4));

  // Q/K/V projections: out = x @ W.T -> W rows are B-matrix columns (native layout)
  dim3 gg(D / 64, (B * S) / 128);
  gemm_bf16_kernel<true><<<gg, 256, 0, stream>>>(xb, wqb, qb, B * S, D, D);
  gemm_bf16_kernel<true><<<gg, 256, 0, stream>>>(xb, wkb, kb, B * S, D, D);
  gemm_bf16_kernel<true><<<gg, 256, 0, stream>>>(xb, wvb, vb, B * S, D, D);

  attn_kernel<<<dim3(S / 128, H, B), 256, 0, stream>>>(qb, kb, vb, cb);

  // output projection -> f32
  gemm_bf16_kernel<false><<<gg, 256, 0, stream>>>(cb, wob, d_out, B * S, D, D);
}
